// Aligner2_747324310049
// MI455X (gfx1250) — matrix operand order repaired and checked by reading, not compiled, not run on hardware
//
#include <hip/hip_runtime.h>
#include <hip/hip_bf16.h>
#include <math.h>

typedef _Float16 v16h __attribute__((ext_vector_type(16)));
typedef _Float16 h8   __attribute__((ext_vector_type(8)));
typedef float    v8f  __attribute__((ext_vector_type(8)));

#define NB    32
#define NT1   512
#define NT2   2048
#define NCH   80
#define NCHP  96       // K padded to 3x32 for wmma f16 (k=32)
#define TEMPC 0.0005f
#define SLOPE 0.3f
#define NEGV  (-1000000000.0f)

// ---------------------------------------------------------------------------
// Generic 1-D conv + leakyReLU.  Whole Cin slab for a 128-wide t-tile in LDS.
// grid = (T/128, B), block = 128, dyn LDS = Cin*(128+K-1)*4 bytes.
// ---------------------------------------------------------------------------
__global__ __launch_bounds__(128)
void conv1d_lrelu(const float* __restrict__ x, const float* __restrict__ w,
                  const float* __restrict__ bias, float* __restrict__ y,
                  int Cin, int Cout, int T, int K, int pad)
{
    extern __shared__ float xs[];            // [Cin][128+K-1]
    const int TILE = 128;
    const int b  = blockIdx.y;
    const int t0 = blockIdx.x * TILE;
    const int W  = TILE + K - 1;

    const int total = Cin * W;
    for (int idx = threadIdx.x; idx < total; idx += blockDim.x) {
        int c  = idx / W;
        int tt = idx - c * W;
        int t  = t0 + tt - pad;
        xs[idx] = (t >= 0 && t < T) ? x[((size_t)b * Cin + c) * T + t] : 0.0f;
    }
    __syncthreads();

    const int t = t0 + threadIdx.x;
    if (t >= T) return;
    for (int o = 0; o < Cout; ++o) {
        float acc = bias[o];
        const float* wo = w + (size_t)o * Cin * K;
        for (int c = 0; c < Cin; ++c) {
            const float* xc = xs + c * W + threadIdx.x;
            const float* wc = wo + c * K;
            for (int kk = 0; kk < K; ++kk)
                acc = fmaf(wc[kk], xc[kk], acc);
        }
        y[((size_t)b * Cout + o) * T + t] = (acc >= 0.0f) ? acc : SLOPE * acc;
    }
}

// ---------------------------------------------------------------------------
// Pack [B,C,T] f32 -> [B,T,96] f16 (zero padded) + per-t squared norm.
// grid = (T/256, B), block = 256.
// ---------------------------------------------------------------------------
__global__ __launch_bounds__(256)
void pack_f16(const float* __restrict__ src, _Float16* __restrict__ dst,
              float* __restrict__ sq, int C, int T)
{
    const int b = blockIdx.y;
    const int t = blockIdx.x * 256 + threadIdx.x;
    if (t >= T) return;
    float s = 0.0f;
    _Float16* drow = dst + ((size_t)b * T + t) * NCHP;
    for (int c = 0; c < NCHP; ++c) {
        float v = 0.0f;
        if (c < C) { v = src[((size_t)b * C + c) * T + t]; s = fmaf(v, v, s); }
        drow[c] = (_Float16)v;
    }
    sq[(size_t)b * T + t] = s;
}

// ---------------------------------------------------------------------------
// Transposed log-prior: plogT[b,t2,t1] = log(prior[b,t1,t2]+1e-8)
// grid = (T2/32, T1/32, B), block = (32,8). 32x32 LDS tile transpose.
// ---------------------------------------------------------------------------
__global__ __launch_bounds__(256)
void transpose_logprior(const float* __restrict__ prior, float* __restrict__ plogT)
{
    __shared__ float tile[32][33];
    const int b = blockIdx.z;
    const int t2_0 = blockIdx.x * 32;
    const int t1_0 = blockIdx.y * 32;
    #pragma unroll
    for (int k = 0; k < 4; ++k) {
        int t1 = t1_0 + threadIdx.y + k * 8;
        int t2 = t2_0 + threadIdx.x;
        tile[threadIdx.y + k * 8][threadIdx.x] =
            __logf(prior[((size_t)b * NT1 + t1) * NT2 + t2] + 1e-8f);
    }
    __syncthreads();
    #pragma unroll
    for (int k = 0; k < 4; ++k) {
        int t2 = t2_0 + threadIdx.y + k * 8;
        int t1 = t1_0 + threadIdx.x;
        plogT[((size_t)b * NT2 + t2) * NT1 + t1] = tile[threadIdx.x][threadIdx.y + k * 8];
    }
}

// ---------------------------------------------------------------------------
// Fused attention: WMMA GEMM (K=96 as 3x v_wmma_f32_16x16x32_f16) -> LDS row
// buffer -> dual online-softmax -> write attn_logprob + attn_soft.
// grid = (T2/64, B), block = 128 (4 waves, 16 rows each),
// dyn LDS = 4*16*512*4 = 128 KB (qk row buffer).
// A-fragment layout (16-bit A 16x32, contiguous k as loaded here): lane L holds M=L&15,
//   elem e -> c = kk*32 + (L>>4)*16 + e (same order as B).
// B-fragment (16-bit B 32x16): lane L holds N=L&15, elem e -> c = kk*32 + (L>>4)*16 + e.
// C/D f32 16x16: lane L -> N=L&15, vgpr r -> M = r + 8*(L>>4).
// ---------------------------------------------------------------------------
__global__ __launch_bounds__(128)
void attn_fused(const _Float16* __restrict__ kh,  const float* __restrict__ k2g,
                const _Float16* __restrict__ qh,  const float* __restrict__ q2g,
                const float* __restrict__ plogT,
                float* __restrict__ out_logprob,  float* __restrict__ out_soft)
{
    extern __shared__ float qkbuf[];                 // [4][16][512]
    const int b    = blockIdx.y;
    const int wave = threadIdx.x >> 5;
    const int lane = threadIdx.x & 31;
    const int m0   = blockIdx.x * 64 + wave * 16;    // first t2 row of this wave
    const int hsel = lane >> 4;
    const int l16  = lane & 15;
    float* myqk = qkbuf + (size_t)wave * 16 * NT1;   // [16][512]

    // ---- A fragments (q rows m0..m0+15, K = 96 channels) ----
    const _Float16* qrow = qh + ((size_t)b * NT2 + (m0 + l16)) * NCHP;
    v16h afrag[3];
    #pragma unroll
    for (int kk = 0; kk < 3; ++kk) {
        afrag[kk] = *(const v16h*)(qrow + kk * 32 + hsel * 16);
    }

    // ---- GEMM over 32 N-tiles of t1 ----
    for (int nt = 0; nt < 32; ++nt) {
        const _Float16* krow = kh + ((size_t)b * NT1 + (nt * 16 + l16)) * NCHP;
        v8f acc = {};
        #pragma unroll
        for (int kk = 0; kk < 3; ++kk) {
            v16h bfrag = *(const v16h*)(krow + kk * 32 + hsel * 16);
            acc = __builtin_amdgcn_wmma_f32_16x16x32_f16(
                false, afrag[kk], false, bfrag, (short)0, acc, false, false);
        }
        #pragma unroll
        for (int r = 0; r < 8; ++r)
            myqk[(size_t)(r + 8 * hsel) * NT1 + (nt * 16 + l16)] = acc[r];
    }
    __syncthreads();

    // ---- k2 per lane-column (reused for all 16 rows) ----
    float k2reg[16];
    #pragma unroll
    for (int jj = 0; jj < 16; ++jj)
        k2reg[jj] = k2g[(size_t)b * NT1 + jj * 32 + lane];

    // ---- per-row dual softmax ----
    for (int mr = 0; mr < 16; ++mr) {
        const int t2 = m0 + mr;
        const float base = -TEMPC * q2g[(size_t)b * NT2 + t2];
        const float* prow = plogT + ((size_t)b * NT2 + t2) * NT1;
        const float* qkrow = myqk + (size_t)mr * NT1;

        float lp[16];
        float m1 = -INFINITY, s1 = 0.0f, m2 = -INFINITY, s2 = 0.0f;
        #pragma unroll
        for (int jj = 0; jj < 16; ++jj) {
            int col  = jj * 32 + lane;
            float v0 = fmaf(2.0f * TEMPC, qkrow[col], base - TEMPC * k2reg[jj]);
            float pl = prow[col];
            lp[jj]   = pl;
            float v2 = v0 + pl;
            float n1 = fmaxf(m1, v0);
            s1 = s1 * __expf(m1 - n1) + __expf(v0 - n1); m1 = n1;
            float n2 = fmaxf(m2, v2);
            s2 = s2 * __expf(m2 - n2) + __expf(v2 - n2); m2 = n2;
        }
        #pragma unroll
        for (int off = 16; off >= 1; off >>= 1) {
            float om1 = __shfl_xor(m1, off, 32), os1 = __shfl_xor(s1, off, 32);
            float nm1 = fmaxf(m1, om1);
            s1 = s1 * __expf(m1 - nm1) + os1 * __expf(om1 - nm1); m1 = nm1;
            float om2 = __shfl_xor(m2, off, 32), os2 = __shfl_xor(s2, off, 32);
            float nm2 = fmaxf(m2, om2);
            s2 = s2 * __expf(m2 - nm2) + os2 * __expf(om2 - nm2); m2 = nm2;
        }
        const float lse1 = m1 + __logf(s1);
        const float lse2 = m2 + __logf(s2);

        float* olog = out_logprob + ((size_t)b * NT2 + t2) * NT1;
        float* osft = out_soft    + ((size_t)b * NT2 + t2) * NT1;
        #pragma unroll
        for (int jj = 0; jj < 16; ++jj) {
            int col  = jj * 32 + lane;
            float v0 = fmaf(2.0f * TEMPC, qkrow[col], base - TEMPC * k2reg[jj]);
            olog[col] = v0 - lse1 + lp[jj];
            osft[col] = __expf(v0 + lp[jj] - lse2);
        }
    }
}

// ---------------------------------------------------------------------------
// MAS: one workgroup per batch, 512 threads (j = t1), DP bits kept in LDS.
// dyn LDS = 2048*16*4 (bits) + 512*4 (lp) + 2048*2 (path) = 137216 B.
// ---------------------------------------------------------------------------
__global__ __launch_bounds__(512)
void mas_kernel(const float* __restrict__ soft, const int* __restrict__ enc_len,
                const int* __restrict__ dec_len, float* __restrict__ hard,
                float* __restrict__ dur)
{
    extern __shared__ char smem[];
    unsigned int*   bits_s = (unsigned int*)smem;                 // [2048][16]
    float*          lp_s   = (float*)(smem + 2048 * 16 * 4);      // [512]
    unsigned short* path_s = (unsigned short*)(smem + 2048 * 16 * 4 + 512 * 4); // [2048]

    const int b   = blockIdx.x;
    const int j   = threadIdx.x;
    const int inl = enc_len[b];
    const int outl = dec_len[b];

    // ---- forward DP ----
    float logp = (j == 0) ? 0.0f : NEGV;
    for (int i = 0; i < NT2; ++i) {
        lp_s[j] = logp;
        __syncthreads();
        float shifted = (j >= 1) ? lp_s[j - 1] : NEGV;
        float sv = soft[((size_t)b * NT2 + i) * NT1 + j];
        float lo = (j < inl) ? __logf(sv) : NEGV;
        if (i == 0 && j > 0) lo = NEGV;
        bool take = (j >= 1) && (shifted >= logp);
        unsigned long long bm = __ballot(take);
        if ((j & 31) == 0) bits_s[i * 16 + (j >> 5)] = (unsigned int)bm;
        logp = lo + (take ? shifted : logp);
        __syncthreads();
    }

    // ---- backtrack (serial, LDS only) ----
    if (j == 0) {
        int curr = inl - 1;
        for (int i = NT2 - 1; i >= 0; --i) {
            path_s[i] = (unsigned short)curr;
            if (i < outl) {
                unsigned int w = bits_s[i * 16 + (curr >> 5)];
                curr -= (int)((w >> (curr & 31)) & 1u);
            }
        }
    }
    __syncthreads();

    // ---- durations ----
    int cnt = 0;
    for (int i = 0; i < outl; ++i)
        if ((int)path_s[i] == j) ++cnt;
    dur[(size_t)b * NT1 + j] = (float)cnt;

    // ---- dense hard alignment (overwrites plog scratch region) ----
    for (int i = 0; i < NT2; ++i) {
        float v = (i < outl && (int)path_s[i] == j) ? 1.0f : 0.0f;
        hard[((size_t)b * NT2 + i) * NT1 + j] = v;
    }
}

// ---------------------------------------------------------------------------
extern "C" void kernel_launch(void* const* d_in, const int* in_sizes, int n_in,
                              void* d_out, int out_size, void* d_ws, size_t ws_size,
                              hipStream_t stream)
{
    const float* enc_in = (const float*)d_in[0];
    const float* dec_in = (const float*)d_in[1];
    const int*   enc_len = (const int*)d_in[2];
    const int*   dec_len = (const int*)d_in[3];
    /* d_in[4] enc_mask unused by the reference */
    const float* prior = (const float*)d_in[5];
    const float* kw1 = (const float*)d_in[6];  const float* kb1 = (const float*)d_in[7];
    const float* kw2 = (const float*)d_in[8];  const float* kb2 = (const float*)d_in[9];
    const float* qw1 = (const float*)d_in[10]; const float* qb1 = (const float*)d_in[11];
    const float* qw2 = (const float*)d_in[12]; const float* qb2 = (const float*)d_in[13];
    const float* qw3 = (const float*)d_in[14]; const float* qb3 = (const float*)d_in[15];

    const size_t PLANE = (size_t)NB * NT2 * NT1;      // 33,554,432
    float* out_logprob = (float*)d_out;
    float* out_soft    = out_logprob + PLANE;
    float* out_hard    = out_logprob + 2 * PLANE;     // also used as plogT scratch
    float* out_dur     = out_logprob + 3 * PLANE;

    // workspace carving (all offsets 256B aligned)
    char* ws = (char*)d_ws;
    size_t o = 0;
    float*    k1buf = (float*)(ws + o);      o += (size_t)NB * NCH * NT1 * 4;
    float*    k2buf = (float*)(ws + o);      o += (size_t)NB * NCH * NT1 * 4;
    float*    qa    = (float*)(ws + o);      o += (size_t)NB * NCH * NT2 * 4;
    float*    qb    = (float*)(ws + o);      o += (size_t)NB * NCH * NT2 * 4;
    _Float16* khp   = (_Float16*)(ws + o);   o += (size_t)NB * NT1 * NCHP * 2;
    float*    k2sum = (float*)(ws + o);      o += (size_t)NB * NT1 * 4;
    _Float16* qhp   = (_Float16*)(ws + o);   o += (size_t)NB * NT2 * NCHP * 2;
    float*    q2sum = (float*)(ws + o);      o += (size_t)NB * NT2 * 4;

    // 1-2. key convs (Cin=512 then 80, K=3, pad=1)
    conv1d_lrelu<<<dim3(NT1 / 128, NB), 128, 512 * 130 * 4, stream>>>(
        enc_in, kw1, kb1, k1buf, 512, NCH, NT1, 3, 1);
    conv1d_lrelu<<<dim3(NT1 / 128, NB), 128, NCH * 130 * 4, stream>>>(
        k1buf, kw2, kb2, k2buf, NCH, NCH, NT1, 3, 1);

    // 3-5. query convs (K=7, pad=3), ping-pong qa/qb
    conv1d_lrelu<<<dim3(NT2 / 128, NB), 128, NCH * 134 * 4, stream>>>(
        dec_in, qw1, qb1, qa, NCH, NCH, NT2, 7, 3);
    conv1d_lrelu<<<dim3(NT2 / 128, NB), 128, NCH * 134 * 4, stream>>>(
        qa, qw2, qb2, qb, NCH, NCH, NT2, 7, 3);
    conv1d_lrelu<<<dim3(NT2 / 128, NB), 128, NCH * 134 * 4, stream>>>(
        qb, qw3, qb3, qa, NCH, NCH, NT2, 7, 3);

    // 6-7. pack to f16 WMMA-friendly layout + squared norms
    pack_f16<<<dim3(NT1 / 256, NB), 256, 0, stream>>>(k2buf, khp, k2sum, NCH, NT1);
    pack_f16<<<dim3(NT2 / 256, NB), 256, 0, stream>>>(qa, qhp, q2sum, NCH, NT2);

    // 8. transposed log-prior into the hard-output region (overwritten by MAS)
    transpose_logprior<<<dim3(NT2 / 32, NT1 / 32, NB), dim3(32, 8), 0, stream>>>(
        prior, out_hard);

    // 9. fused WMMA attention + dual softmax
    attn_fused<<<dim3(NT2 / 64, NB), 128, 4 * 16 * NT1 * 4, stream>>>(
        khp, k2sum, qhp, q2sum, out_hard, out_logprob, out_soft);

    // 10. MAS (reads soft, overwrites hard region, writes durations)
    mas_kernel<<<dim3(NB), 512, 2048 * 16 * 4 + 512 * 4 + 2048 * 2, stream>>>(
        out_soft, enc_len, dec_len, out_hard, out_dur);
}
